// NoCrossLayer_51170240364935
// MI455X (gfx1250) — matrix operand order repaired and checked by reading, not compiled, not run on hardware
//
#include <hip/hip_runtime.h>
#include <hip/hip_bf16.h>
#include <cstdint>

// ---------- types ----------
typedef __bf16 bf16x16 __attribute__((ext_vector_type(16)));
typedef float  f32x8   __attribute__((ext_vector_type(8)));
typedef float  f32x4   __attribute__((ext_vector_type(4)));
typedef unsigned int u32x4 __attribute__((ext_vector_type(4)));

union Frag { bf16x16 v; u32x4 q[2]; };

#define BATCH 4
#define NPTS  4096
#define DIM   64
#define KNN_K 16
#define LEAKY 0.1f

__device__ inline unsigned short f2bf(float x) {
  unsigned int u = __float_as_uint(x);
  u += 0x7FFFu + ((u >> 16) & 1u);      // round-to-nearest-even
  return (unsigned short)(u >> 16);
}
__device__ inline unsigned int pack2(float a, float b) {
  return (unsigned int)f2bf(a) | ((unsigned int)f2bf(b) << 16);
}

// ---------- kernel 1: [B,D,N] -> [B,N,D] tiled transpose ----------
__global__ __launch_bounds__(256) void transposeDN(const float* __restrict__ in,
                                                   float* __restrict__ out) {
  __shared__ float tile[32][33];
  const int b  = blockIdx.z;
  const int c0 = blockIdx.y * 32;
  const int n0 = blockIdx.x * 32;
  const int tx = threadIdx.x, ty = threadIdx.y;
  const float* src = in  + (size_t)b * DIM * NPTS;
  float*       dst = out + (size_t)b * NPTS * DIM;
  #pragma unroll
  for (int j = 0; j < 32; j += 8)
    tile[ty + j][tx] = src[(size_t)(c0 + ty + j) * NPTS + n0 + tx];
  __syncthreads();
  #pragma unroll
  for (int j = 0; j < 32; j += 8)
    dst[(size_t)(n0 + ty + j) * DIM + c0 + tx] = tile[tx][ty + j];
}

// ---------- kernel 2: weight swizzle into WMMA B-operand fragments ----------
// 44 fragments: L0: f=ks*4+t (ks<5,t<4), L1: 20+ks*4+t (ks<2), L2: 28+ks*8+t (ks<2,t<8)
// fragment f, lane l, dword j holds bf16 pair (k0,k0+1) of column n:
//   n = t*16 + (l&15);  k0 = ks*32 + ((l&16)?8:0) + 2*j + (j>=4?8:0)
__global__ __launch_bounds__(256) void wswizzle(const float* __restrict__ W0,
                                                const float* __restrict__ W1,
                                                const float* __restrict__ W2,
                                                unsigned int* __restrict__ wf) {
  for (int i = threadIdx.x; i < 44 * 256; i += 256) {
    const int f = i >> 8, r = i & 255, lane = r >> 3, j = r & 7;
    int ks, t, indim;
    const float* W;
    if (f < 20)      { ks = f >> 2;        t = f & 3;        W = W0; indim = 131; }
    else if (f < 28) { ks = (f - 20) >> 2; t = (f - 20) & 3; W = W1; indim = 64;  }
    else             { ks = (f - 28) >> 3; t = (f - 28) & 7; W = W2; indim = 64;  }
    const int nOut = t * 16 + (lane & 15);
    const int k0   = ks * 32 + ((lane & 16) ? 8 : 0) + 2 * j + ((j >= 4) ? 8 : 0);
    const float a = (k0     < indim) ? W[nOut * indim + k0]     : 0.f;
    const float b = (k0 + 1 < indim) ? W[nOut * indim + k0 + 1] : 0.f;
    wf[i] = pack2(a, b);
  }
}

// ---------- kernel 3: brute-force KNN (top-16 by squared distance) ----------
__global__ __launch_bounds__(64) void knn_kernel(const float* __restrict__ pc1,
                                                 const float* __restrict__ pc2,
                                                 int* __restrict__ idxb) {
  __shared__ float cx[256], cy[256], cz[256];
  const int b = blockIdx.x >> 6;
  const int n = ((blockIdx.x & 63) << 6) + threadIdx.x;
  const float* p1 = pc1 + (size_t)b * 3 * NPTS;
  const float* p2 = pc2 + (size_t)b * 3 * NPTS;
  const float qx = p1[n], qy = p1[NPTS + n], qz = p1[2 * NPTS + n];

  float d[KNN_K]; int id[KNN_K];
  #pragma unroll
  for (int k = 0; k < KNN_K; ++k) { d[k] = 3.4e38f; id[k] = 0; }

  for (int tile = 0; tile < NPTS / 256; ++tile) {
    const int m0 = tile << 8;
    __syncthreads();
    for (int l = threadIdx.x; l < 256; l += 64) {
      cx[l] = p2[m0 + l];
      cy[l] = p2[NPTS + m0 + l];
      cz[l] = p2[2 * NPTS + m0 + l];
    }
    __syncthreads();
    for (int j = 0; j < 256; ++j) {
      const float dx = cx[j] - qx, dy = cy[j] - qy, dz = cz[j] - qz;
      const float dist = dx * dx + dy * dy + dz * dz;
      if (dist < d[KNN_K - 1]) {            // rare path: ordered insert
        float nd = dist; int ni = m0 + j;
        #pragma unroll
        for (int k = 0; k < KNN_K; ++k) {
          const bool s = nd < d[k];
          const float td = d[k]; const int ti = id[k];
          d[k]  = s ? nd : d[k];  id[k] = s ? ni : id[k];
          nd    = s ? td : nd;    ni    = s ? ti : ni;
        }
      }
    }
  }
  int* o = idxb + ((size_t)b * NPTS + n) * KNN_K;
  #pragma unroll
  for (int k = 0; k < KNN_K; ++k) o[k] = id[k];
}

// ---------- WMMA helpers ----------
__device__ inline f32x8 wmma_bf16(Frag a, Frag b, f32x8 c) {
  return __builtin_amdgcn_wmma_f32_16x16x32_bf16(false, a.v, false, b.v,
                                                 (short)0, c, false, false);
}
// A-operand (16x32 bf16): lane<16 -> row M=lane, K {kb..kb+7, kb+16..kb+23}
__device__ inline Frag loadA(const uint16_t (*X)[168], int lane, int colbase, int kb) {
  Frag a;
  const uint16_t* p = &X[lane & 15][colbase + kb + ((lane & 16) ? 8 : 0)];
  a.q[0] = *(const u32x4*)p;
  a.q[1] = *(const u32x4*)(p + 16);
  return a;
}
__device__ inline Frag loadB(const unsigned int* wf, int f, int lane) {
  Frag b;
  const u32x4* p = (const u32x4*)(wf + ((f << 5) + lane) * 8);
  b.q[0] = p[0]; b.q[1] = p[1];
  return b;
}
// C/D 16x16 f32: VGPR j, lane l -> M = j + (l<16?0:8), N = l&15
__device__ inline void storeH(uint16_t (*X)[168], int lane, int colbase, int tile,
                              f32x8 acc, const float* sBias, int boff) {
  const int c  = tile * 16 + (lane & 15);
  const float bias = sBias[boff + c];
  const int rb = (lane & 16) ? 8 : 0;
  #pragma unroll
  for (int j = 0; j < 8; ++j) {
    float v = acc[j] + bias;
    v = (v >= 0.f) ? v : LEAKY * v;
    X[rb + j][colbase + c] = f2bf(v);
  }
}

// ---------- kernel 4: gather + 3-layer WMMA MLP + neighbor-max ----------
__global__ __launch_bounds__(256) void mlp_kernel(
    const float* __restrict__ pc1, const float* __restrict__ pc2,
    const float* __restrict__ f1t, const float* __restrict__ f2t,
    const int* __restrict__ idxb, const unsigned int* __restrict__ wf,
    const float* __restrict__ b0, const float* __restrict__ b1,
    const float* __restrict__ b2, float* __restrict__ out) {
  __shared__ __align__(16) uint16_t sX[8][16][168];  // per-wave A-matrix / H buffers
  __shared__ float sBias[256];
  __shared__ float sOut[8][128];

  const int tid = threadIdx.x;
  const int w = tid >> 5, lane = tid & 31;

  sBias[tid] = (tid < 64) ? b0[tid] : (tid < 128) ? b1[tid - 64] : b2[tid - 128];
  __syncthreads();

  const int q = blockIdx.x * 8 + w;           // global query id
  const int b = q >> 12, n = q & 4095;
  uint16_t (*X)[168] = sX[w];

  // zero K-padding region (cols 128..167)
  for (int i = lane; i < 320; i += 32) {
    const int r = i / 20, c = 128 + (i % 20) * 2;
    *(uint32_t*)&X[r][c] = 0u;
  }
  const int row = lane & 15, half = lane >> 4;
  const int m = idxb[q * KNN_K + row];
  const f32x4* g1 = (const f32x4*)(f1t + (((size_t)b << 12) + n) * DIM + half * 32);
  const f32x4* g2 = (const f32x4*)(f2t + (((size_t)b << 12) + m) * DIM + half * 32);
  #pragma unroll
  for (int j = 0; j < 8; ++j) {
    const f32x4 v1 = g1[j], v2 = g2[j];
    const int c = half * 32 + j * 4;
    *(uint32_t*)&X[row][c]          = pack2(v1.x, v1.y);
    *(uint32_t*)&X[row][c + 2]      = pack2(v1.z, v1.w);
    *(uint32_t*)&X[row][64 + c]     = pack2(v2.x, v2.y);
    *(uint32_t*)&X[row][64 + c + 2] = pack2(v2.z, v2.w);
  }
  if (lane < 16) {
    const float* p1 = pc1 + (size_t)b * 3 * NPTS;
    const float* p2 = pc2 + (size_t)b * 3 * NPTS;
    X[row][128] = f2bf(p2[m]            - p1[n]);
    X[row][129] = f2bf(p2[NPTS + m]     - p1[NPTS + n]);
    X[row][130] = f2bf(p2[2*NPTS + m]   - p1[2*NPTS + n]);
  }
  __builtin_amdgcn_wave_barrier();   // LDS in-order per wave; keep compiler honest

  // ---- layer 0: 16x160 @ 160x64 -> HA (cols 0..63) ----
  f32x8 a0 = {0,0,0,0,0,0,0,0}, a1 = a0, a2 = a0, a3 = a0;
  #pragma unroll
  for (int ks = 0; ks < 5; ++ks) {
    const Frag A = loadA(X, lane, 0, ks * 32);
    a0 = wmma_bf16(A, loadB(wf, ks * 4 + 0, lane), a0);
    a1 = wmma_bf16(A, loadB(wf, ks * 4 + 1, lane), a1);
    a2 = wmma_bf16(A, loadB(wf, ks * 4 + 2, lane), a2);
    a3 = wmma_bf16(A, loadB(wf, ks * 4 + 3, lane), a3);
  }
  __builtin_amdgcn_wave_barrier();
  storeH(X, lane, 0, 0, a0, sBias, 0);
  storeH(X, lane, 0, 1, a1, sBias, 0);
  storeH(X, lane, 0, 2, a2, sBias, 0);
  storeH(X, lane, 0, 3, a3, sBias, 0);
  __builtin_amdgcn_wave_barrier();

  // ---- layer 1: 16x64 @ 64x64 -> HB (cols 88..151) ----
  f32x8 c0 = {0,0,0,0,0,0,0,0}, c1 = c0, c2 = c0, c3 = c0;
  #pragma unroll
  for (int ks = 0; ks < 2; ++ks) {
    const Frag A = loadA(X, lane, 0, ks * 32);
    c0 = wmma_bf16(A, loadB(wf, 20 + ks * 4 + 0, lane), c0);
    c1 = wmma_bf16(A, loadB(wf, 20 + ks * 4 + 1, lane), c1);
    c2 = wmma_bf16(A, loadB(wf, 20 + ks * 4 + 2, lane), c2);
    c3 = wmma_bf16(A, loadB(wf, 20 + ks * 4 + 3, lane), c3);
  }
  __builtin_amdgcn_wave_barrier();
  storeH(X, lane, 88, 0, c0, sBias, 64);
  storeH(X, lane, 88, 1, c1, sBias, 64);
  storeH(X, lane, 88, 2, c2, sBias, 64);
  storeH(X, lane, 88, 3, c3, sBias, 64);
  __builtin_amdgcn_wave_barrier();

  // ---- layer 2: 16x64 @ 64x128 -> bias+leaky -> max over 16 neighbors ----
  #pragma unroll
  for (int th = 0; th < 2; ++th) {
    f32x8 e0 = {0,0,0,0,0,0,0,0}, e1 = e0, e2 = e0, e3 = e0;
    #pragma unroll
    for (int ks = 0; ks < 2; ++ks) {
      const Frag A = loadA(X, lane, 88, ks * 32);
      e0 = wmma_bf16(A, loadB(wf, 28 + ks * 8 + th * 4 + 0, lane), e0);
      e1 = wmma_bf16(A, loadB(wf, 28 + ks * 8 + th * 4 + 1, lane), e1);
      e2 = wmma_bf16(A, loadB(wf, 28 + ks * 8 + th * 4 + 2, lane), e2);
      e3 = wmma_bf16(A, loadB(wf, 28 + ks * 8 + th * 4 + 3, lane), e3);
    }
    f32x8 accs[4] = {e0, e1, e2, e3};
    #pragma unroll
    for (int tt = 0; tt < 4; ++tt) {
      const int tile = th * 4 + tt;
      const int c = tile * 16 + (lane & 15);
      const float bias = sBias[128 + c];
      float mx = -3.4e38f;
      #pragma unroll
      for (int j = 0; j < 8; ++j) {
        float v = accs[tt][j] + bias;
        v = (v >= 0.f) ? v : LEAKY * v;
        mx = fmaxf(mx, v);
      }
      mx = fmaxf(mx, __shfl_xor(mx, 16, 32));  // combine M<8 with M>=8 halves
      if ((lane & 16) == 0) sOut[w][c] = mx;
    }
  }

  __syncthreads();
  // coalesced output: out[b][c][n0..n0+7]
  const int bb = blockIdx.x >> 9;
  const int n0 = (blockIdx.x & 511) * 8;
  for (int i = tid; i < 1024; i += 256) {
    const int c = i >> 3, j = i & 7;
    out[(((size_t)bb * 128) + c) * NPTS + n0 + j] = sOut[j][c];
  }
}

// ---------- launch ----------
extern "C" void kernel_launch(void* const* d_in, const int* in_sizes, int n_in,
                              void* d_out, int out_size, void* d_ws, size_t ws_size,
                              hipStream_t stream) {
  const float* pc1   = (const float*)d_in[0];
  const float* pc2   = (const float*)d_in[1];
  const float* feat1 = (const float*)d_in[2];
  const float* feat2 = (const float*)d_in[3];
  const float* W0    = (const float*)d_in[4];
  const float* b0    = (const float*)d_in[5];
  const float* W1    = (const float*)d_in[6];
  const float* b1    = (const float*)d_in[7];
  const float* W2    = (const float*)d_in[8];
  const float* b2    = (const float*)d_in[9];
  float* out = (float*)d_out;

  float* f1t = (float*)d_ws;                                   // 4 MB
  float* f2t = f1t + (size_t)BATCH * NPTS * DIM;               // 4 MB
  int*   idxb = (int*)(f2t + (size_t)BATCH * NPTS * DIM);      // 1 MB
  unsigned int* wf = (unsigned int*)(idxb + (size_t)BATCH * NPTS * KNN_K); // 44 KB

  transposeDN<<<dim3(NPTS / 32, DIM / 32, BATCH), dim3(32, 8), 0, stream>>>(feat1, f1t);
  transposeDN<<<dim3(NPTS / 32, DIM / 32, BATCH), dim3(32, 8), 0, stream>>>(feat2, f2t);
  wswizzle<<<1, 256, 0, stream>>>(W0, W1, W2, wf);
  knn_kernel<<<BATCH * 64, 64, 0, stream>>>(pc1, pc2, idxb);
  mlp_kernel<<<(BATCH * NPTS) / 8, 256, 0, stream>>>(pc1, pc2, f1t, f2t, idxb, wf,
                                                     b0, b1, b2, out);
}
